// Backbone_33827162423729
// MI455X (gfx1250) — matrix operand order repaired and checked by reading, not compiled, not run on hardware
//
#include <hip/hip_runtime.h>

// ---------------------------------------------------------------------------
// DGCNN-style backbone for MI455X (gfx1250, wave32, WMMA).
// All GEMM-like work (kNN inner products, edge-conv linear, tail linear) goes
// through v_wmma_f32_16x16x32_bf16 with 64x64 block tiles (4 WMMAs per wave
// per K-step, shared B fragment). BN stats via two-pass recompute + atomics.
// ---------------------------------------------------------------------------

typedef __attribute__((ext_vector_type(16))) __bf16         v16bf;
typedef __attribute__((ext_vector_type(8)))  float          v8f;
typedef __attribute__((ext_vector_type(8)))  unsigned short u16x8;
typedef __attribute__((ext_vector_type(16))) unsigned short u16x16;

constexpr int BATCH = 4;
constexpr int NPTS  = 1024;
constexpr int KNN   = 20;
constexpr float EPSV      = 1e-5f;
constexpr float NEG_SLOPE = 0.2f;

// LDS tile stride (in halves). 40 halves = 80 bytes, 16B-aligned rows.
constexpr int TST = 40;

__device__ __forceinline__ unsigned short f2bf_bits(float f) {
    unsigned u = __float_as_uint(f);
    unsigned r = u + 0x7FFFu + ((u >> 16) & 1u);   // RNE
    return (unsigned short)(r >> 16);
}

__device__ __forceinline__ unsigned encodeF(float f) {
    unsigned u = __float_as_uint(f);
    return (u & 0x80000000u) ? ~u : (u | 0x80000000u);
}
__device__ __forceinline__ float decodeF(unsigned e) {
    unsigned u = (e & 0x80000000u) ? (e & 0x7FFFFFFFu) : ~e;
    return __uint_as_float(u);
}

// Build a v16bf fragment from two aligned 8-half LDS chunks.
__device__ __forceinline__ v16bf frag16(const unsigned short* p0,
                                        const unsigned short* p1) {
    u16x8 lo = *(const u16x8*)p0;
    u16x8 hi = *(const u16x8*)p1;
    u16x16 c = __builtin_shufflevector(lo, hi,
                 0,1,2,3,4,5,6,7,8,9,10,11,12,13,14,15);
    return __builtin_bit_cast(v16bf, c);
}

// A-fragment (16-bit A 16x32 ISA layout): row = M within tile, chunks at
// k = {0,16} for lanes 0-15 and k = {8,24} for lanes 16-31.
__device__ __forceinline__ v16bf afrag(const unsigned short* tile,
                                       int rowBase, int lane) {
    int row = rowBase + (lane & 15);
    int a0  = (lane < 16) ? 0 : 8;
    return frag16(&tile[row * TST + a0], &tile[row * TST + a0 + 16]);
}

// B-fragment (16-bit B 32x16 ISA layout): row = N (col of output),
// chunks at k = {0,16} for lanes 0-15 and k = {8,24} for lanes 16-31.
__device__ __forceinline__ v16bf bfrag(const unsigned short* tile,
                                       int rowBase, int lane) {
    int row = rowBase + (lane & 15);
    int b0  = (lane < 16) ? 0 : 8;
    return frag16(&tile[row * TST + b0], &tile[row * TST + b0 + 16]);
}

// ---------------------------------------------------------------------------
// sq[b*N+n] = sum_c x[b,c,n]^2
// ---------------------------------------------------------------------------
__global__ void sqnorm_kernel(const float* __restrict__ X, int ldx, int C,
                              float* __restrict__ sq) {
    int t = blockIdx.x * blockDim.x + threadIdx.x;
    if (t >= BATCH * NPTS) return;
    int b = t / NPTS, n = t - b * NPTS;
    const float* p = X + ((size_t)b * ldx) * NPTS + n;
    float s = 0.f;
    for (int c = 0; c < C; ++c) { float v = p[(size_t)c * NPTS]; s += v * v; }
    sq[t] = s;
}

// ---------------------------------------------------------------------------
// neg_dist[b,i,j] = 2*<x_i,x_j> - |x_i|^2 - |x_j|^2   via bf16 WMMA
// block = 128 threads (4 waves); tile = 64 i x 64 j; wave w owns j-strip w.
// ---------------------------------------------------------------------------
__global__ void dist_kernel(const float* __restrict__ X, int ldx, int C,
                            const float* __restrict__ sq,
                            float* __restrict__ dist) {
    int b     = blockIdx.z;
    int iBase = blockIdx.x * 64;
    int jBase = blockIdx.y * 64;

    __shared__ __align__(16) unsigned short sA[64 * TST];
    __shared__ __align__(16) unsigned short sB[64 * TST];

    int tid  = threadIdx.x;
    int wave = tid >> 5;
    int lane = tid & 31;

    v8f acc[4];
    for (int s = 0; s < 4; ++s)
        for (int r = 0; r < 8; ++r) acc[s][r] = 0.f;

    int kTiles = (C + 31) >> 5;
    const float* Xb = X + ((size_t)b * ldx) * NPTS;

    for (int kt = 0; kt < kTiles; ++kt) {
        for (int e = tid; e < 64 * 32; e += 128) {
            int r = e >> 5, kl = e & 31, c = (kt << 5) + kl;
            float va = (c < C) ? Xb[(size_t)c * NPTS + (iBase + r)] : 0.f;
            float vb = (c < C) ? Xb[(size_t)c * NPTS + (jBase + r)] : 0.f;
            sA[r * TST + kl] = f2bf_bits(va);
            sB[r * TST + kl] = f2bf_bits(vb);
        }
        __syncthreads();
        v16bf bf = bfrag(sB, wave << 4, lane);
        for (int s = 0; s < 4; ++s) {
            v16bf af = afrag(sA, s << 4, lane);
            acc[s] = __builtin_amdgcn_wmma_f32_16x16x32_bf16(
                         false, af, false, bf, (short)0, acc[s], false, false);
        }
        __syncthreads();
    }

    int col   = lane & 15;
    int mrow0 = (lane < 16) ? 0 : 8;
    int j     = jBase + (wave << 4) + col;
    float sqj = sq[b * NPTS + j];
    for (int s = 0; s < 4; ++s) {
        for (int r = 0; r < 8; ++r) {
            int i = iBase + (s << 4) + mrow0 + r;
            float nd = 2.f * acc[s][r] - sq[b * NPTS + i] - sqj;
            dist[((size_t)b * NPTS + i) * NPTS + j] = nd;
        }
    }
}

// ---------------------------------------------------------------------------
// top-k (k=20) per row of neg_dist; iterative masked argmax (stable, lowest
// index wins ties, matching lax.top_k ordering).
// ---------------------------------------------------------------------------
__global__ void topk_kernel(const float* __restrict__ dist,
                            int* __restrict__ idx) {
    int row = blockIdx.x;                      // b*N + i
    const float* d = dist + (size_t)row * NPTS;
    __shared__ float vals[NPTS];
    __shared__ float rv[256];
    __shared__ int   ri[256];
    int tid = threadIdx.x;
    for (int j = tid; j < NPTS; j += 256) vals[j] = d[j];
    __syncthreads();
    for (int t = 0; t < KNN; ++t) {
        float best = -3.4e38f; int bi = 0;
        for (int j = tid; j < NPTS; j += 256) {
            float v = vals[j];
            if (v > best) { best = v; bi = j; }
        }
        rv[tid] = best; ri[tid] = bi;
        __syncthreads();
        for (int s = 128; s > 0; s >>= 1) {
            if (tid < s) {
                if (rv[tid + s] > rv[tid] ||
                    (rv[tid + s] == rv[tid] && ri[tid + s] < ri[tid])) {
                    rv[tid] = rv[tid + s]; ri[tid] = ri[tid + s];
                }
            }
            __syncthreads();
        }
        if (tid == 0) { idx[row * KNN + t] = ri[0]; vals[ri[0]] = -3.4e38f; }
        __syncthreads();
    }
}

// ---------------------------------------------------------------------------
// A-tile value generator for the fused GEMMs.
// mode 1 (edge conv): row m -> (b,n,kk); value c<C: x[b,c,idx]-x[b,c,n]
//                                        c>=C: x[b,c-C,n]      (Ktot = 2C)
// mode 0 (linear)   : row m -> (b,n);    value x[b,c,n]        (Ktot = C)
// ---------------------------------------------------------------------------
__device__ __forceinline__ float a_value(const float* __restrict__ X, int ldx,
                                         const int* __restrict__ idx,
                                         int C, int Ktot, int mode,
                                         int m, int c) {
    if (c >= Ktot) return 0.f;
    if (mode == 1) {
        int bb  = m / (NPTS * KNN);
        int rem = m - bb * (NPTS * KNN);
        int n   = rem / KNN;
        int j   = idx[m];
        const float* Xb = X + ((size_t)bb * ldx) * NPTS;
        if (c < C) return Xb[(size_t)c * NPTS + j] - Xb[(size_t)c * NPTS + n];
        return Xb[(size_t)(c - C) * NPTS + n];
    } else {
        int bb = m / NPTS, n = m - bb * NPTS;
        return X[((size_t)bb * ldx + c) * NPTS + n];
    }
}

// Shared tile loop body for both GEMM passes: stages a 64(M)x32(K) A tile and
// a 64(O)x32(K) B tile, runs 4 WMMAs per wave (one B frag, 4 A frags).
#define GEMM_TILE_LOOP()                                                      \
    for (int kt = 0; kt < kTiles; ++kt) {                                     \
        for (int e = tid; e < 64 * 32; e += 128) {                            \
            int r = e >> 5, kl = e & 31, c = (kt << 5) + kl;                  \
            sA[r * TST + kl] =                                                \
                f2bf_bits(a_value(X, ldx, idx, C, Ktot, mode, mBase + r, c)); \
            float wv = (c < Ktot) ? W[(size_t)(oBase + r) * Ktot + c] : 0.f;  \
            sB[r * TST + kl] = f2bf_bits(wv);                                 \
        }                                                                     \
        __syncthreads();                                                      \
        v16bf bf = bfrag(sB, wave << 4, lane);                                \
        for (int s = 0; s < 4; ++s) {                                         \
            v16bf af = afrag(sA, s << 4, lane);                               \
            acc[s] = __builtin_amdgcn_wmma_f32_16x16x32_bf16(                 \
                         false, af, false, bf, (short)0, acc[s],              \
                         false, false);                                       \
        }                                                                     \
        __syncthreads();                                                      \
    }

// Pass 1: gather + WMMA GEMM, accumulate per-channel sum / sumsq (BN stats).
__global__ void gemm_stats_kernel(const float* __restrict__ X, int ldx,
                                  const int* __restrict__ idx,
                                  const float* __restrict__ W,
                                  int C, int Ktot, int mode,
                                  float* __restrict__ sums,
                                  float* __restrict__ sumsq) {
    int mBase = blockIdx.x * 64;
    int oBase = blockIdx.y * 64;
    __shared__ __align__(16) unsigned short sA[64 * TST];
    __shared__ __align__(16) unsigned short sB[64 * TST];
    int tid = threadIdx.x, wave = tid >> 5, lane = tid & 31;

    v8f acc[4];
    for (int s = 0; s < 4; ++s)
        for (int r = 0; r < 8; ++r) acc[s][r] = 0.f;
    int kTiles = (Ktot + 31) >> 5;

    GEMM_TILE_LOOP();

    float s1 = 0.f, s2 = 0.f;
    for (int s = 0; s < 4; ++s)
        for (int r = 0; r < 8; ++r) { float v = acc[s][r]; s1 += v; s2 += v * v; }
    s1 += __shfl_down(s1, 16);
    s2 += __shfl_down(s2, 16);
    if (lane < 16) {
        int o = oBase + (wave << 4) + lane;
        atomicAdd(&sums[o], s1);
        atomicAdd(&sumsq[o], s2);
    }
}

// Pass 2: recompute tile, BN + leakyReLU, then either max-over-k via encoded
// atomicMax (mode 1) or direct store (mode 0).
__global__ void gemm_apply_kernel(const float* __restrict__ X, int ldx,
                                  const int* __restrict__ idx,
                                  const float* __restrict__ W,
                                  int C, int Ktot, int mode,
                                  const float* __restrict__ meanv,
                                  const float* __restrict__ rstdv,
                                  const float* __restrict__ gamma,
                                  const float* __restrict__ beta,
                                  float* __restrict__ outF,
                                  unsigned* __restrict__ outU, int ldo) {
    int mBase = blockIdx.x * 64;
    int oBase = blockIdx.y * 64;
    __shared__ __align__(16) unsigned short sA[64 * TST];
    __shared__ __align__(16) unsigned short sB[64 * TST];
    int tid = threadIdx.x, wave = tid >> 5, lane = tid & 31;

    v8f acc[4];
    for (int s = 0; s < 4; ++s)
        for (int r = 0; r < 8; ++r) acc[s][r] = 0.f;
    int kTiles = (Ktot + 31) >> 5;

    GEMM_TILE_LOOP();

    int col = lane & 15;
    int o   = oBase + (wave << 4) + col;
    float mn = meanv[o], rs = rstdv[o], g = gamma[o], bt = beta[o];
    int mrow0 = (lane < 16) ? 0 : 8;
    for (int s = 0; s < 4; ++s) {
        for (int r = 0; r < 8; ++r) {
            int m = mBase + (s << 4) + mrow0 + r;
            float v = (acc[s][r] - mn) * rs * g + bt;
            v = (v > 0.f) ? v : v * NEG_SLOPE;
            if (mode == 1) {
                int bb  = m / (NPTS * KNN);
                int rem = m - bb * (NPTS * KNN);
                int n   = rem / KNN;
                atomicMax(&outU[((size_t)bb * ldo + o) * NPTS + n], encodeF(v));
            } else {
                int bb = m / NPTS, n = m - bb * NPTS;
                outF[((size_t)bb * ldo + o) * NPTS + n] = v;
            }
        }
    }
}

// ---------------------------------------------------------------------------
// small helpers
// ---------------------------------------------------------------------------
__global__ void zero_stats_kernel(float* sums, float* sumsq) {
    int t = threadIdx.x;            // 512 threads
    sums[t] = 0.f; sumsq[t] = 0.f;
}

__global__ void finalize_stats_kernel(const float* sums, const float* sumsq,
                                      int O, float cnt,
                                      float* meanv, float* rstdv) {
    int o = threadIdx.x;
    if (o >= O) return;
    float mn  = sums[o] / cnt;
    float var = sumsq[o] / cnt - mn * mn;
    meanv[o] = mn;
    rstdv[o] = rsqrtf(var + EPSV);
}

__global__ void init_out_kernel(unsigned* out, int O, int ldo) {
    int t = blockIdx.x * blockDim.x + threadIdx.x;
    if (t >= BATCH * O * NPTS) return;
    int b = t / (O * NPTS), rem = t - b * O * NPTS;
    int o = rem / NPTS, n = rem - o * NPTS;
    out[((size_t)b * ldo + o) * NPTS + n] = 0u;   // < encode(any float)
}

__global__ void decode_out_kernel(unsigned* out, int O, int ldo) {
    int t = blockIdx.x * blockDim.x + threadIdx.x;
    if (t >= BATCH * O * NPTS) return;
    int b = t / (O * NPTS), rem = t - b * O * NPTS;
    int o = rem / NPTS, n = rem - o * NPTS;
    size_t a = ((size_t)b * ldo + o) * NPTS + n;
    out[a] = __float_as_uint(decodeF(out[a]));
}

__global__ void final_linear_kernel(const float* __restrict__ X,
                                    const float* __restrict__ W,
                                    const float* __restrict__ bias,
                                    float* __restrict__ out) {
    int t = blockIdx.x * blockDim.x + threadIdx.x;
    if (t >= BATCH * 3 * NPTS) return;
    int n = t % NPTS, rem = t / NPTS;
    int o = rem % 3, b = rem / 3;
    float s = bias[o];
    const float* Xb = X + ((size_t)b * 64) * NPTS + n;
    for (int c = 0; c < 64; ++c) s += Xb[(size_t)c * NPTS] * W[o * 64 + c];
    out[((size_t)b * 3 + o) * NPTS + n] = s;
}

// ---------------------------------------------------------------------------
// host orchestration
// ---------------------------------------------------------------------------
struct Ws {
    float* feat; float* dist; float* sq; int* idx;
    float* sums; float* sumsq; float* meanv; float* rstdv;
    float* bufA; float* bufB; float* xcat;
};

static void run_edge_conv(const float* x, int C, int ldx,
                          float* out, int O, int ldo,
                          const float* W, const float* g, const float* bt,
                          const Ws& w, hipStream_t s) {
    sqnorm_kernel<<<(BATCH * NPTS + 255) / 256, 256, 0, s>>>(x, ldx, C, w.sq);
    dim3 gd(NPTS / 64, NPTS / 64, BATCH);
    dist_kernel<<<gd, 128, 0, s>>>(x, ldx, C, w.sq, w.dist);
    topk_kernel<<<BATCH * NPTS, 256, 0, s>>>(w.dist, w.idx);

    zero_stats_kernel<<<1, 512, 0, s>>>(w.sums, w.sumsq);
    int Mtot = BATCH * NPTS * KNN;
    dim3 g1(Mtot / 64, O / 64);
    gemm_stats_kernel<<<g1, 128, 0, s>>>(x, ldx, w.idx, W, C, 2 * C, 1,
                                         w.sums, w.sumsq);
    finalize_stats_kernel<<<1, 512, 0, s>>>(w.sums, w.sumsq, O, (float)Mtot,
                                            w.meanv, w.rstdv);
    int ne = BATCH * O * NPTS;
    init_out_kernel<<<(ne + 255) / 256, 256, 0, s>>>((unsigned*)out, O, ldo);
    gemm_apply_kernel<<<g1, 128, 0, s>>>(x, ldx, w.idx, W, C, 2 * C, 1,
                                         w.meanv, w.rstdv, g, bt,
                                         nullptr, (unsigned*)out, ldo);
    decode_out_kernel<<<(ne + 255) / 256, 256, 0, s>>>((unsigned*)out, O, ldo);
}

static void run_linear_bn(const float* x, int C, int ldx,
                          float* out, int O,
                          const float* W, const float* g, const float* bt,
                          const Ws& w, hipStream_t s) {
    zero_stats_kernel<<<1, 512, 0, s>>>(w.sums, w.sumsq);
    int Mtot = BATCH * NPTS;
    dim3 g1(Mtot / 64, O / 64);
    gemm_stats_kernel<<<g1, 128, 0, s>>>(x, ldx, nullptr, W, C, C, 0,
                                         w.sums, w.sumsq);
    finalize_stats_kernel<<<1, 512, 0, s>>>(w.sums, w.sumsq, O, (float)Mtot,
                                            w.meanv, w.rstdv);
    gemm_apply_kernel<<<g1, 128, 0, s>>>(x, ldx, nullptr, W, C, C, 0,
                                         w.meanv, w.rstdv, g, bt,
                                         out, nullptr, O);
}

extern "C" void kernel_launch(void* const* d_in, const int* in_sizes, int n_in,
                              void* d_out, int out_size, void* d_ws, size_t ws_size,
                              hipStream_t stream) {
    (void)in_sizes; (void)n_in; (void)out_size; (void)ws_size;
    const float* xin[2] = { (const float*)d_in[0], (const float*)d_in[1] };
    const float* P[32];
    for (int i = 0; i < 32; ++i) P[i] = (const float*)d_in[2 + i];
    // params, in setup_inputs() insertion order:
    // 0..14  : enc{i}_{W,g,b} i=0..4
    // 15..17 : tail_W0, tail_g0, tail_b0
    // 18..23 : tail_ec{0,1}_{W,g,b}
    // 24..29 : dec_ec{0,1}_{W,g,b}
    // 30,31  : dec_W, dec_bias

    float* f = (float*)d_ws;
    Ws w;
    size_t off = 0;
    w.feat  = f + off; off += (size_t)BATCH * 1024 * NPTS;   // 4M floats
    w.dist  = f + off; off += (size_t)BATCH * NPTS * NPTS;   // 4M floats
    w.sq    = f + off; off += (size_t)BATCH * NPTS;
    w.idx   = (int*)(f + off); off += (size_t)BATCH * NPTS * KNN;
    w.sums  = f + off; off += 512;
    w.sumsq = f + off; off += 512;
    w.meanv = f + off; off += 512;
    w.rstdv = f + off; off += 512;
    w.bufA  = f + off; off += (size_t)BATCH * 512 * NPTS;
    w.bufB  = f + off; off += (size_t)BATCH * 512 * NPTS;
    w.xcat  = f + off; off += (size_t)BATCH * 256 * NPTS;

    const int encO[5]   = { 64, 64, 128, 256, 512 };
    const int encOfs[5] = { 0, 64, 128, 256, 512 };

    for (int sIdx = 0; sIdx < 2; ++sIdx) {
        const float* cur = xin[sIdx];
        int curC = 3, curLD = 3;
        for (int i = 0; i < 5; ++i) {
            float* o = w.feat + (size_t)encOfs[i] * NPTS;   // slice, ld=1024
            run_edge_conv(cur, curC, curLD, o, encO[i], 1024,
                          P[3 * i], P[3 * i + 1], P[3 * i + 2], w, stream);
            cur = o; curC = encO[i]; curLD = 1024;
        }
        // tail: linear 1024->512 + bn1d + leaky
        run_linear_bn(w.feat, 1024, 1024, w.bufA, 512,
                      P[15], P[16], P[17], w, stream);
        // tail edge convs: 512->256, 256->128 (into xcat half)
        run_edge_conv(w.bufA, 512, 512, w.bufB, 256, 256,
                      P[18], P[19], P[20], w, stream);
        run_edge_conv(w.bufB, 256, 256,
                      w.xcat + (size_t)sIdx * 128 * NPTS, 128, 256,
                      P[21], P[22], P[23], w, stream);
    }

    // decoder edge convs on concat(x1,x2): 256->128, 128->64
    run_edge_conv(w.xcat, 256, 256, w.bufA, 128, 128,
                  P[24], P[25], P[26], w, stream);
    run_edge_conv(w.bufA, 128, 128, w.bufB, 64, 64,
                  P[27], P[28], P[29], w, stream);

    // final linear 64 -> 3 (+bias)
    final_linear_kernel<<<(BATCH * 3 * NPTS + 255) / 256, 256, 0, stream>>>(
        w.bufB, P[30], P[31], (float*)d_out);
}
